// RobustSum_16544214024900
// MI455X (gfx1250) — compile-verified, matrix operand order checked by reading, not run on hardware
//
#include <hip/hip_runtime.h>

#define S_DIM 2048
#define D_DIM 512
#define B_DIM 4
#define EPS_W 0.01f
#define L1_EPS 1e-12f

typedef __bf16 bf16;
typedef __attribute__((ext_vector_type(16))) __bf16 v16bf;
typedef __attribute__((ext_vector_type(8)))  __bf16 v8bf;
typedef __attribute__((ext_vector_type(8)))  float   v8f;

// ---- WMMA helpers -------------------------------------------------------

// Load one 16x32 bf16 operand tile (A-layout, or B-layout from the
// transposed matrix): lane owns row (lane&15); lane group g=lane>>4 reads
// K = {g*8..g*8+7} and {16+g*8..16+g*8+7} as two 16B contiguous chunks.
__device__ __forceinline__ v16bf ld16(const bf16* p) {
  v8bf a = *(const v8bf*)(p);
  v8bf b = *(const v8bf*)(p + 16);
  return __builtin_shufflevector(a, b, 0,1,2,3,4,5,6,7,8,9,10,11,12,13,14,15);
}

__device__ __forceinline__ v8f wmma_bf(v16bf a, v16bf b, v8f c) {
  return __builtin_amdgcn_wmma_f32_16x16x32_bf16(false, a, false, b, (short)0, c, false, false);
}

// split-3 product: (ahi+alo)*(bhi+blo) ~= ahi*bhi + ahi*blo + alo*bhi
__device__ __forceinline__ v8f wmma3(v16bf ahi, v16bf alo, v16bf bhi, v16bf blo, v8f c) {
  c = wmma_bf(ahi, bhi, c);
  c = wmma_bf(ahi, blo, c);
  c = wmma_bf(alo, bhi, c);
  return c;
}

__device__ __forceinline__ void splitf(float x, bf16& h, bf16& l) {
  h = (bf16)x;
  l = (bf16)(x - (float)h);
}

__device__ __forceinline__ float red16(float v) {
  v += __shfl_xor(v, 1, 32);
  v += __shfl_xor(v, 2, 32);
  v += __shfl_xor(v, 4, 32);
  v += __shfl_xor(v, 8, 32);
  return v;
}

// ---- prep kernels -------------------------------------------------------

// V (B,S,D) fp32 -> bf16 splits (B,S,D) and transposed splits (B,D,S)
__global__ __launch_bounds__(256) void split_transpose_kernel(
    const float* __restrict__ V, bf16* __restrict__ Vhi, bf16* __restrict__ Vlo,
    bf16* __restrict__ Vthi, bf16* __restrict__ Vtlo) {
  __shared__ float th[32][33];
  __shared__ float tl[32][33];
  int b = blockIdx.z;
  int d0 = blockIdx.x * 32, s0 = blockIdx.y * 32;
  size_t bo = (size_t)b * S_DIM * D_DIM;
#pragma unroll
  for (int i = 0; i < 4; ++i) {
    int sl = threadIdx.y + i * 8;
    int s = s0 + sl, d = d0 + threadIdx.x;
    float x = V[bo + (size_t)s * D_DIM + d];
    bf16 h, l; splitf(x, h, l);
    size_t o = bo + (size_t)s * D_DIM + d;
    Vhi[o] = h; Vlo[o] = l;
    th[sl][threadIdx.x] = (float)h;
    tl[sl][threadIdx.x] = (float)l;
  }
  __syncthreads();
#pragma unroll
  for (int i = 0; i < 4; ++i) {
    int dl = threadIdx.y + i * 8;
    int d = d0 + dl, s = s0 + threadIdx.x;
    size_t o = bo + (size_t)d * S_DIM + s;
    Vthi[o] = (bf16)th[threadIdx.x][dl];
    Vtlo[o] = (bf16)tl[threadIdx.x][dl];
  }
}

// v2[b*S + j] = ||V[b,j,:]||^2 ; one wave per row
__global__ __launch_bounds__(256) void rownorm_kernel(const float* __restrict__ V,
                                                      float* __restrict__ v2) {
  int wid = threadIdx.x >> 5, lane = threadIdx.x & 31;
  int row = blockIdx.x * 8 + wid;
  const float* p = V + (size_t)row * D_DIM;
  float s = 0.f;
#pragma unroll
  for (int i = 0; i < D_DIM / 32; ++i) { float x = p[lane + i * 32]; s += x * x; }
  s += __shfl_xor(s, 16, 32);
  s = red16(s);
  if (lane == 0) v2[row] = s;
}

// ---- GEMM to M:  M = op(A) @ V  (N = D = 512, K = S = 2048) -------------
// AFP32: A-operand is fp32 (the input A matrix), split in-register.
// USE_L1: scale output row i by 1/max(l1[i],1e-12).
// FINAL: write fp32 output only; else write bf16 splits + row-norm atomics.
template <bool AFP32, bool USE_L1, bool FINAL>
__global__ __launch_bounds__(256) void gemm_kernel(
    const bf16* __restrict__ Ahi, const bf16* __restrict__ Alo,
    const float* __restrict__ Afp,
    const bf16* __restrict__ Bthi, const bf16* __restrict__ Btlo,
    const float* __restrict__ l1,
    float* __restrict__ Mout, bf16* __restrict__ Mhi, bf16* __restrict__ Mlo,
    float* __restrict__ m2out) {
  const int b = blockIdx.z;
  const size_t bSS = (size_t)S_DIM * S_DIM, bSD = (size_t)S_DIM * D_DIM;
  if (AFP32) { Afp += (size_t)b * bSS; } else { Ahi += (size_t)b * bSS; Alo += (size_t)b * bSS; }
  Bthi += (size_t)b * bSD; Btlo += (size_t)b * bSD;
  if (USE_L1) l1 += (size_t)b * S_DIM;
  if (FINAL) { Mout += (size_t)b * bSD; }
  else { Mhi += (size_t)b * bSD; Mlo += (size_t)b * bSD; m2out += (size_t)b * S_DIM; }

  const int wid = threadIdx.x >> 5, lane = threadIdx.x & 31;
  const int r15 = lane & 15, g = lane >> 4;
  const int row0 = blockIdx.y * 128 + (wid >> 2) * 64;
  const int col0 = blockIdx.x * 256 + (wid & 3) * 64;

  v8f zero = {0.f, 0.f, 0.f, 0.f, 0.f, 0.f, 0.f, 0.f};
  v8f acc[4][4];
#pragma unroll
  for (int m = 0; m < 4; ++m)
#pragma unroll
    for (int n = 0; n < 4; ++n) acc[m][n] = zero;

#pragma unroll 1
  for (int kc = 0; kc < S_DIM; kc += 32) {
    v16bf ahi[4], alo[4];
#pragma unroll
    for (int mt = 0; mt < 4; ++mt) {
      size_t off = (size_t)(row0 + mt * 16 + r15) * S_DIM + kc + g * 8;
      if (AFP32) {
        const float* p = Afp + off;
        float fa[16];
        *(float4*)(fa + 0)  = *(const float4*)(p);
        *(float4*)(fa + 4)  = *(const float4*)(p + 4);
        *(float4*)(fa + 8)  = *(const float4*)(p + 16);
        *(float4*)(fa + 12) = *(const float4*)(p + 20);
#pragma unroll
        for (int i = 0; i < 16; ++i) { bf16 h, l; splitf(fa[i], h, l); ahi[mt][i] = h; alo[mt][i] = l; }
      } else {
        ahi[mt] = ld16(Ahi + off);
        alo[mt] = ld16(Alo + off);
      }
    }
#pragma unroll
    for (int nt = 0; nt < 4; ++nt) {
      size_t boff = (size_t)(col0 + nt * 16 + r15) * S_DIM + kc + g * 8;
      v16bf bhi = ld16(Bthi + boff);
      v16bf blo = ld16(Btlo + boff);
#pragma unroll
      for (int mt = 0; mt < 4; ++mt) acc[mt][nt] = wmma3(ahi[mt], alo[mt], bhi, blo, acc[mt][nt]);
    }
  }

  // epilogue
#pragma unroll
  for (int mt = 0; mt < 4; ++mt) {
    float sc[8];
    if (USE_L1) {
#pragma unroll
      for (int vr = 0; vr < 8; ++vr)
        sc[vr] = 1.0f / fmaxf(l1[row0 + mt * 16 + vr + g * 8], L1_EPS);
    }
    float rs[8] = {0.f, 0.f, 0.f, 0.f, 0.f, 0.f, 0.f, 0.f};
#pragma unroll
    for (int nt = 0; nt < 4; ++nt) {
      int col = col0 + nt * 16 + r15;
#pragma unroll
      for (int vr = 0; vr < 8; ++vr) {
        int row = row0 + mt * 16 + vr + g * 8;
        float val = acc[mt][nt][vr];
        if (USE_L1) val *= sc[vr];
        size_t o = (size_t)row * D_DIM + col;
        if (FINAL) {
          Mout[o] = val;
        } else {
          bf16 h, l; splitf(val, h, l);
          Mhi[o] = h; Mlo[o] = l;
          rs[vr] += val * val;
        }
      }
    }
    if (!FINAL) {
#pragma unroll
      for (int vr = 0; vr < 8; ++vr) {
        float s = red16(rs[vr]);
        if (r15 == 0) unsafeAtomicAdd(&m2out[row0 + mt * 16 + vr + g * 8], s);
      }
    }
  }
}

// ---- weights: XY = M @ V^T, then ww = A / (dist+eps), store splits + l1 --
__global__ __launch_bounds__(256) void weights_kernel(
    const bf16* __restrict__ Mhi, const bf16* __restrict__ Mlo, const float* __restrict__ m2,
    const bf16* __restrict__ Vhi, const bf16* __restrict__ Vlo, const float* __restrict__ v2,
    const float* __restrict__ A, bf16* __restrict__ Whi, bf16* __restrict__ Wlo,
    float* __restrict__ l1out) {
  const int b = blockIdx.z;
  const size_t bSS = (size_t)S_DIM * S_DIM, bSD = (size_t)S_DIM * D_DIM;
  Mhi += (size_t)b * bSD; Mlo += (size_t)b * bSD;
  Vhi += (size_t)b * bSD; Vlo += (size_t)b * bSD;
  m2 += (size_t)b * S_DIM; v2 += (size_t)b * S_DIM; l1out += (size_t)b * S_DIM;
  A += (size_t)b * bSS; Whi += (size_t)b * bSS; Wlo += (size_t)b * bSS;

  const int wid = threadIdx.x >> 5, lane = threadIdx.x & 31;
  const int r15 = lane & 15, g = lane >> 4;
  const int row0 = blockIdx.y * 128 + (wid >> 2) * 64;
  const int col0 = blockIdx.x * 256 + (wid & 3) * 64;

  v8f zero = {0.f, 0.f, 0.f, 0.f, 0.f, 0.f, 0.f, 0.f};
  v8f acc[4][4];
#pragma unroll
  for (int m = 0; m < 4; ++m)
#pragma unroll
    for (int n = 0; n < 4; ++n) acc[m][n] = zero;

#pragma unroll 1
  for (int kc = 0; kc < D_DIM; kc += 32) {
    v16bf ahi[4], alo[4];
#pragma unroll
    for (int mt = 0; mt < 4; ++mt) {
      size_t off = (size_t)(row0 + mt * 16 + r15) * D_DIM + kc + g * 8;
      ahi[mt] = ld16(Mhi + off);
      alo[mt] = ld16(Mlo + off);
    }
#pragma unroll
    for (int nt = 0; nt < 4; ++nt) {
      // B = V^T  =>  B^T = V (row-major, ld = D): row owner is column j
      size_t boff = (size_t)(col0 + nt * 16 + r15) * D_DIM + kc + g * 8;
      v16bf bhi = ld16(Vhi + boff);
      v16bf blo = ld16(Vlo + boff);
#pragma unroll
      for (int mt = 0; mt < 4; ++mt) acc[mt][nt] = wmma3(ahi[mt], alo[mt], bhi, blo, acc[mt][nt]);
    }
  }

  // epilogue: dist -> w -> ww, write splits, accumulate l1 row sums
#pragma unroll
  for (int mt = 0; mt < 4; ++mt) {
    float m2r[8];
#pragma unroll
    for (int vr = 0; vr < 8; ++vr) m2r[vr] = m2[row0 + mt * 16 + vr + g * 8];
    float rs[8] = {0.f, 0.f, 0.f, 0.f, 0.f, 0.f, 0.f, 0.f};
#pragma unroll
    for (int nt = 0; nt < 4; ++nt) {
      int col = col0 + nt * 16 + r15;
      float v2c = v2[col];
#pragma unroll
      for (int vr = 0; vr < 8; ++vr) {
        int row = row0 + mt * 16 + vr + g * 8;
        float xy = acc[mt][nt][vr];
        float d2 = fmaxf(m2r[vr] + v2c - 2.0f * xy, 0.0f);
        float ww = A[(size_t)row * S_DIM + col] / (sqrtf(d2) + EPS_W);
        rs[vr] += fabsf(ww);
        size_t o = (size_t)row * S_DIM + col;
        bf16 h, l; splitf(ww, h, l);
        Whi[o] = h; Wlo[o] = l;
      }
    }
#pragma unroll
    for (int vr = 0; vr < 8; ++vr) {
      float s = red16(rs[vr]);
      if (r15 == 0) unsafeAtomicAdd(&l1out[row0 + mt * 16 + vr + g * 8], s);
    }
  }
}

// ---- driver -------------------------------------------------------------

extern "C" void kernel_launch(void* const* d_in, const int* in_sizes, int n_in,
                              void* d_out, int out_size, void* d_ws, size_t ws_size,
                              hipStream_t stream) {
  const float* A = (const float*)d_in[0];  // (B,S,S)
  const float* V = (const float*)d_in[1];  // (B,S,D)
  float* out = (float*)d_out;              // (B,S,D)

  char* w = (char*)d_ws;
  auto carve = [&](size_t n) { char* p = w; w += (n + 255) & ~(size_t)255; return p; };
  const size_t svd2 = (size_t)B_DIM * S_DIM * D_DIM * sizeof(bf16);
  const size_t sss2 = (size_t)B_DIM * S_DIM * S_DIM * sizeof(bf16);
  const size_t sf   = (size_t)B_DIM * S_DIM * sizeof(float);

  bf16* Vhi  = (bf16*)carve(svd2);
  bf16* Vlo  = (bf16*)carve(svd2);
  bf16* Vthi = (bf16*)carve(svd2);
  bf16* Vtlo = (bf16*)carve(svd2);
  bf16* Mhi  = (bf16*)carve(svd2);
  bf16* Mlo  = (bf16*)carve(svd2);
  bf16* Whi  = (bf16*)carve(sss2);
  bf16* Wlo  = (bf16*)carve(sss2);
  float* v2  = (float*)carve(sf);
  float* m2  = (float*)carve(sf);
  float* l1  = (float*)carve(sf);

  const dim3 gU(D_DIM / 256, S_DIM / 128, B_DIM);  // update-style GEMM (N=512)
  const dim3 gW(S_DIM / 256, S_DIM / 128, B_DIM);  // weights GEMM (N=2048)

  split_transpose_kernel<<<dim3(D_DIM / 32, S_DIM / 32, B_DIM), dim3(32, 8), 0, stream>>>(
      V, Vhi, Vlo, Vthi, Vtlo);
  rownorm_kernel<<<dim3(B_DIM * S_DIM / 8), dim3(256), 0, stream>>>(V, v2);

  // M0 = A @ V
  hipMemsetAsync(m2, 0, sf, stream);
  gemm_kernel<true, false, false><<<gU, 256, 0, stream>>>(
      nullptr, nullptr, A, Vthi, Vtlo, nullptr, nullptr, Mhi, Mlo, m2);

  for (int it = 0; it < 3; ++it) {
    hipMemsetAsync(l1, 0, sf, stream);
    weights_kernel<<<gW, 256, 0, stream>>>(Mhi, Mlo, m2, Vhi, Vlo, v2, A, Whi, Wlo, l1);
    if (it < 2) {
      hipMemsetAsync(m2, 0, sf, stream);
      gemm_kernel<false, true, false><<<gU, 256, 0, stream>>>(
          Whi, Wlo, nullptr, Vthi, Vtlo, l1, nullptr, Mhi, Mlo, m2);
    } else {
      gemm_kernel<false, true, true><<<gU, 256, 0, stream>>>(
          Whi, Wlo, nullptr, Vthi, Vtlo, l1, out, nullptr, nullptr, nullptr);
    }
  }
}
